// RNNBatchEnsembleLayer_66116726554735
// MI455X (gfx1250) — matrix operand order repaired and checked by reading, not compiled, not run on hardware
//
#include <hip/hip_runtime.h>

// RNNBatchEnsemble for MI455X (gfx1250, wave32, WMMA f16 16x16x32, f32 acc).
// B=64, S=256, E=4, I=H=512.

#define B_ 64
#define S_ 256
#define E_ 4
#define I_ 512
#define H_ 512

typedef __attribute__((ext_vector_type(16))) _Float16 v16h;
typedef __attribute__((ext_vector_type(8)))  _Float16 v8h;
typedef __attribute__((ext_vector_type(8)))  float    v8f;

// ---------------------------------------------------------------------------
// Fragment loaders in contiguous k order (k = 16*(lane/16) + s), same for A and B.
// A (16x32 f16, row-major, leading dim ld):
//   lanes 0-15  (m=lane):    a[0..15]=A[m][0..15]
//   lanes 16-31 (m=lane-16): a[0..15]=A[m][16..31]
__device__ __forceinline__ v16h load_a16x32(const _Float16* tile, int ld, int lane) {
    int m  = lane & 15;
    int hf = lane >> 4;
    const _Float16* row = tile + (size_t)m * ld;
    v8h lo = *(const v8h*)(row + hf * 16);
    v8h hi = *(const v8h*)(row + hf * 16 + 8);
    v16h a;
#pragma unroll
    for (int t = 0; t < 8; ++t) { a[t] = lo[t]; a[8 + t] = hi[t]; }
    return a;
}

// B (32x16 f16) where B[k][n] = W[ncol_base + n][k0 + k]  (W row-major, ld):
//   lanes 0-15:  b[0..15] = K = k0 + 0..15  of column n = lane
//   lanes 16-31: b[0..15] = K = k0 + 16..31 of column n = lane-16
// -> one contiguous 16-half (32B) run per lane along W's row.
__device__ __forceinline__ v16h load_bT32x16(const _Float16* w, int ld,
                                             int ncol_base, int k0, int lane) {
    int n  = lane & 15;
    int hf = lane >> 4;
    const _Float16* row = w + (size_t)(ncol_base + n) * ld + k0 + hf * 16;
    v8h lo = *(const v8h*)(row);
    v8h hi = *(const v8h*)(row + 8);
    v16h b;
#pragma unroll
    for (int t = 0; t < 8; ++t) { b[t] = lo[t]; b[8 + t] = hi[t]; }
    return b;
}

__device__ __forceinline__ float tanh_fast(float x) {
    float ax = fabsf(x);
    float e  = __expf(-2.0f * ax);          // v_exp_f32 path
    float r  = (1.0f - e) / (1.0f + e);
    return copysignf(r, x);
}

// ---------------------------------------------------------------------------
// Conversion kernels
__global__ void cvt_x_kernel(const float* __restrict__ x, _Float16* __restrict__ x16, int n) {
    for (int i = blockIdx.x * blockDim.x + threadIdx.x; i < n; i += gridDim.x * blockDim.x)
        x16[i] = (_Float16)x[i];
}

// We[e][h][i] = W_ih[h][i] * r[e][i]  (fold ensemble input scale into weights)
__global__ void build_we_kernel(const float* __restrict__ wih, const float* __restrict__ r,
                                _Float16* __restrict__ we16) {
    const int n = E_ * H_ * I_;
    for (int i = blockIdx.x * blockDim.x + threadIdx.x; i < n; i += gridDim.x * blockDim.x) {
        int e   = i / (H_ * I_);
        int rem = i - e * (H_ * I_);
        int h   = rem / I_;
        int ii  = rem - h * I_;
        we16[i] = (_Float16)(wih[h * I_ + ii] * r[e * I_ + ii]);
    }
}

__global__ void cvt_whh_kernel(const float* __restrict__ whh, _Float16* __restrict__ w16) {
    const int n = H_ * H_;
    for (int i = blockIdx.x * blockDim.x + threadIdx.x; i < n; i += gridDim.x * blockDim.x)
        w16[i] = (_Float16)whh[i];
}

// ---------------------------------------------------------------------------
// Input projection: inp[m][e][h] = sum_i x16[m][i] * We16[e][h][i]
// m = b*S + s (16384 rows). Written f32 straight into d_out at its final
// (b,s,e,h) slot; the recurrence reads it there and overwrites in place.
// Grid (8, 256, 4): 64x64 tiles per WG, 4 waves, wave = 16 rows x 64 cols.
__global__ __launch_bounds__(128) void inp_gemm_kernel(const _Float16* __restrict__ x16,
                                                       const _Float16* __restrict__ we16,
                                                       float* __restrict__ out) {
    const int lane    = threadIdx.x & 31;
    const int w       = threadIdx.x >> 5;
    const int colbase = blockIdx.x * 64;
    const int rowbase = blockIdx.y * 64 + w * 16;
    const int e       = blockIdx.z;
    const _Float16* wbase = we16 + (size_t)e * H_ * I_;

    v8f acc[4];
#pragma unroll
    for (int nt = 0; nt < 4; ++nt)
#pragma unroll
        for (int q = 0; q < 8; ++q) acc[nt][q] = 0.0f;

    for (int k0 = 0; k0 < I_; k0 += 32) {
        v16h a = load_a16x32(x16 + (size_t)rowbase * I_ + k0, I_, lane);
#pragma unroll
        for (int nt = 0; nt < 4; ++nt) {
            v16h b = load_bT32x16(wbase, I_, colbase + nt * 16, k0, lane);
            acc[nt] = __builtin_amdgcn_wmma_f32_16x16x32_f16(
                false, a, false, b, (short)0, acc[nt], false, false);
        }
    }

    const int n  = lane & 15;
    const int hf = lane >> 4;
#pragma unroll
    for (int nt = 0; nt < 4; ++nt) {
#pragma unroll
        for (int j = 0; j < 8; ++j) {
            int m   = rowbase + j + 8 * hf;            // row in [0, B*S)
            int col = colbase + nt * 16 + n;
            out[((size_t)m * E_ + e) * H_ + col] = acc[nt][j];
        }
    }
}

// ---------------------------------------------------------------------------
// Recurrence: rows R = b*E + e (256 total). Each WG owns 16 rows x ALL 512
// columns -> fully self-contained across the 256 sequential steps (only an
// in-WG barrier per step). 16 waves; wave w computes columns [32w, 32w+32).
// h state double-buffered in LDS as f16 (A-matrix source via ds_load_b128).
__global__ __launch_bounds__(512) void rnn_rec_kernel(const _Float16* __restrict__ whh16,
                                                      const float* __restrict__ s,
                                                      float* __restrict__ out) {
    __shared__ _Float16 hbuf[2][16][H_];   // 32 KB
    __shared__ float    s_lds[E_][H_];     //  8 KB

    const int tid   = threadIdx.x;
    const int lane  = tid & 31;
    const int w     = tid >> 5;            // 0..15
    const int R0    = blockIdx.x * 16;     // first (b,e) row of this WG
    const int gbase = w * 32;              // 2 N-tiles per wave

    for (int i = tid; i < 16 * H_; i += 512) (&hbuf[0][0][0])[i] = (_Float16)0.0f;
    for (int i = tid; i < E_ * H_; i += 512) (&s_lds[0][0])[i] = s[i];
    __syncthreads();

    const int n  = lane & 15;
    const int hf = lane >> 4;
    const size_t hid_base = (size_t)B_ * S_ * E_ * H_;

    for (int t = 0; t < S_; ++t) {
        const int cur = t & 1, nxt = cur ^ 1;

        v8f acc[2];
#pragma unroll
        for (int nt = 0; nt < 2; ++nt)
#pragma unroll
            for (int q = 0; q < 8; ++q) acc[nt][q] = 0.0f;

        for (int k0 = 0; k0 < H_; k0 += 32) {
            v16h a = load_a16x32(&hbuf[cur][0][k0], H_, lane);   // ds_load_b128
#pragma unroll
            for (int nt = 0; nt < 2; ++nt) {
                v16h b = load_bT32x16(whh16, H_, gbase + nt * 16, k0, lane);
                acc[nt] = __builtin_amdgcn_wmma_f32_16x16x32_f16(
                    false, a, false, b, (short)0, acc[nt], false, false);
            }
        }

#pragma unroll
        for (int nt = 0; nt < 2; ++nt) {
            const int g = gbase + nt * 16 + n;
#pragma unroll
            for (int j = 0; j < 8; ++j) {
                int R = R0 + j + 8 * hf;
                int b = R >> 2;            // E_ == 4
                int e = R & 3;
                size_t addr = ((size_t)(b * S_ + t) * E_ + e) * H_ + g;
                float pre = out[addr] + acc[nt][j];          // inp term (in place)
                float hv  = tanh_fast(pre * s_lds[e][g]);
                out[addr] = hv;                               // outputs[b,t,e,g]
                hbuf[nxt][j + 8 * hf][g] = (_Float16)hv;
                if (t + 1 < S_)
                    __builtin_prefetch(&out[addr + (size_t)E_ * H_], 0, 0);
                else
                    out[hid_base + ((size_t)e * B_ + b) * H_ + g] = hv;  // hidden[0,e,b,h]
            }
        }
        __syncthreads();   // nxt fully written; cur fully consumed
    }
}

// ---------------------------------------------------------------------------
extern "C" void kernel_launch(void* const* d_in, const int* in_sizes, int n_in,
                              void* d_out, int out_size, void* d_ws, size_t ws_size,
                              hipStream_t stream) {
    (void)in_sizes; (void)n_in; (void)out_size; (void)ws_size;
    const float* x   = (const float*)d_in[0];   // (B,S,I)
    const float* Wih = (const float*)d_in[1];   // (H,I)
    const float* Whh = (const float*)d_in[2];   // (H,H)
    const float* r   = (const float*)d_in[3];   // (E,I)
    const float* s   = (const float*)d_in[4];   // (E,H)
    float* out = (float*)d_out;                 // outputs (B,S,E,H) ++ hidden (1,E,B,H)

    char* ws = (char*)d_ws;
    _Float16* x16   = (_Float16*)ws;                                            // 16.8 MB
    _Float16* we16  = (_Float16*)(ws + (size_t)B_ * S_ * I_ * 2);               //  2.1 MB
    _Float16* whh16 = (_Float16*)(ws + (size_t)B_ * S_ * I_ * 2
                                     + (size_t)E_ * H_ * I_ * 2);               //  0.5 MB

    cvt_x_kernel  <<<4096, 256, 0, stream>>>(x, x16, B_ * S_ * I_);
    build_we_kernel<<<2048, 256, 0, stream>>>(Wih, r, we16);
    cvt_whh_kernel<<<512,  256, 0, stream>>>(Whh, whh16);

    inp_gemm_kernel<<<dim3(8, 256, 4), 128, 0, stream>>>(x16, we16, out);
    rnn_rec_kernel <<<16, 512, 0, stream>>>(whh16, s, out);
}
